// GGRU_49761491091551
// MI455X (gfx1250) — matrix operand order repaired and checked by reading, not compiled, not run on hardware
//
#include <hip/hip_runtime.h>
#include <hip/hip_bf16.h>
#include <math.h>

// ---------------------------------------------------------------------------
// GGRU block for MI455X (gfx1250): bf16 WMMA GEMMs + fused epilogues.
// Wave32, v_wmma_f32_16x16x32_bf16 for every contraction.
// Fragment-ordered LDS tiles (32B contiguous per lane), double-buffered with
// one barrier per K-step; staging uses GLOBAL_LOAD_ASYNC_TO_LDS_B128 when the
// toolchain exposes it, else b128 load + b128 LDS store.
// ---------------------------------------------------------------------------

typedef __attribute__((ext_vector_type(16))) __bf16       v16bf;
typedef __attribute__((ext_vector_type(8)))  float        v8f;
typedef __attribute__((ext_vector_type(4)))  unsigned int u32x4;
typedef __attribute__((ext_vector_type(4)))  int          v4i_;

union Pk8 { u32x4 u; __bf16 h[8]; };

#if defined(__HIP_DEVICE_COMPILE__) && \
    __has_builtin(__builtin_amdgcn_global_load_async_to_lds_b128) && \
    __has_builtin(__builtin_amdgcn_s_wait_asynccnt)
#define HAVE_ASYNC_LDS 1
#else
#define HAVE_ASYNC_LDS 0
#endif

#if HAVE_ASYNC_LDS
__device__ __forceinline__ void async_b128(const __bf16* gsrc, __bf16* ldst) {
    __builtin_amdgcn_global_load_async_to_lds_b128(
        (__attribute__((address_space(1))) v4i_*)gsrc,
        (__attribute__((address_space(3))) v4i_*)ldst, 0, 0);
}
#endif

__device__ __forceinline__ void wait_async_then_sync() {
#if HAVE_ASYNC_LDS
    __builtin_amdgcn_s_wait_asynccnt(0);
#endif
    __syncthreads();
}

#define NB    32
#define CCH   64
#define TTT   256
#define VVV   25
#define ICN   16
#define NSUB  3
#define TV    6400     // T*V
#define NCTV  409600   // C*T*V per n

// ---------------- workspace layout (bytes) ----------------
#define OFF_HBF   ((size_t)0)            // 32*64*6400 bf16 = 26,214,400
#define OFF_FEAT  ((size_t)26214400)     // 26,214,400
#define OFF_AB    ((size_t)52428800)     // 32*96*6400 bf16 = 39,321,600
#define OFF_Z     ((size_t)91750400)     // 96*16384*25 bf16 = 78,643,200
#define OFF_GCN   ((size_t)170393600)    // 26,214,400
#define OFF_MSG   ((size_t)196608000)    // 26,214,400
#define OFF_DH    ((size_t)222822400)    // 32*6400*192 bf16 = 78,643,200
#define OFF_LOG   ((size_t)301465600)    // 96*1024 f32 = 393,216
#define OFF_ATT   ((size_t)301858816)    // 96*1024 bf16 = 196,608
#define OFF_SMALL ((size_t)302055424)    // packed small weights (padded 256KB)
#define OFF_ABT   ((size_t)302317568)    // aT [96][25][4096] bf16 = 19,660,800
#define OFF_BBT   ((size_t)321978368)    // bT [96][25][4096] bf16 = 19,660,800
#define OFF_DI    OFF_AB                 // alias: ab+z are dead when Di is built

// small-region offsets (bytes within OFF_SMALL)
#define SW_WAB  0        // bf16 [96][64]
#define SW_CDW  12288    // bf16 [64][192]
#define SW_TCW  36864    // bf16 [64][576]
#define SW_WI   110592   // bf16 [64][192]
#define SW_WH   135168   // bf16 [64][192]
#define SW_ABB  159744   // f32 [96]
#define SW_CDB  160128   // f32 [64]
#define SW_IB   160384   // f32 [192]

// ---------------------------------------------------------------------------
struct GemmArgs {
    const __bf16* A;
    const __bf16* B;
    void*         D;
    int  M, Nn, K;
    int  sAm, sAk;  long long aOuter; int aInner; int aDiv;
    int  sBk, sBn;  long long bOuter; int bInner; int bDiv;
    int  ldD;       long long dOuter;
    const float* bias;
};

// Fragment-ordered LDS tiles:
//   A 16x32 bf16 tile: lane = hiK*16 + (m&15), e = 8*(k>=16) + (k-16*(k>=16)-8*hiK),
//     hiK = (k>>3)&1. An 8-aligned k-chunk at fixed m -> contiguous e (b128 store).
//   B 32x16 bf16 tile: lane = ((k>>3)&1)*16 + (n&15), e = 8*(k>>4) + (k&7).
//     An 8-aligned k-chunk at fixed n -> contiguous e (b128 store).
//
// APATH: 0 = sAk==1 (k-contiguous rows), 1 = sAm==1 (m-contiguous cols)
// BPATH: 0 = sBn==1 (n-contiguous rows), 1 = sBk==1 (k-contiguous cols)
// BIASMODE: 0 none, 1 per-row, 2 per-col.  OUTBF16: D dtype.
template<int APATH, bool AALIGN, int BPATH, int BIASMODE, bool OUTBF16>
__global__ __launch_bounds__(256, 2)
void gemm_bf16_wmma(GemmArgs ga) {
    __shared__ __bf16 AsF[2][4][32][16];   // [buf][tileRow][lane][e]
    __shared__ __bf16 BsF[2][4][32][16];   // [buf][tileCol][lane][e]

    const int tid  = threadIdx.x;
    const int wave = tid >> 5;
    const int lane = tid & 31;
    const int hi   = lane >> 4;
    const int l15  = lane & 15;

    const int bz = blockIdx.z;
    const long long aOff = (long long)(bz / ga.aDiv) * ga.aOuter
                         + (long long)(bz % ga.aDiv) * ga.aInner;
    const long long bOff = (long long)(bz / ga.bDiv) * ga.bOuter
                         + (long long)(bz % ga.bDiv) * ga.bInner;
    const long long dOff = (long long)bz * ga.dOuter;

    const int blockM = blockIdx.y * 64;
    const int blockN = blockIdx.x * 64;
    const int tr  = wave >> 1;
    const int tc0 = (wave & 1) * 2;
    const int tc1 = tc0 + 1;

    // ---- loop-invariant staging coordinates ----
    // A path0: thread = (row, k-chunk)
    const int rA  = tid >> 2;
    const int c8A = (tid & 3) << 3;
    const int gmA = blockM + rA;
    // A path1: thread = (k-col, m-chunk)
    const int colA = tid >> 3;
    const int m8A  = (tid & 7) << 3;
    const int gm0A = blockM + m8A;
    // B path0: thread = (k-row, n-chunk)
    const int rowB = tid >> 3;
    const int c8B  = (tid & 7) << 3;
    // B path1: thread = (n-col, k-chunk)
    const int k8B = (tid >> 6) << 3;
    const int cB  = tid & 63;
    const int gnB = blockN + cB;

    auto loadA = [&](int kb, Pk8& pk, bool& asy, int buf) {
        if constexpr (APATH == 0) {
            const __bf16* src = ga.A + aOff + (long long)gmA * ga.sAm + (kb + c8A);
            __bf16* dst = &AsF[buf][rA >> 4]
                              [(((c8A >> 3) & 1) << 4) + (rA & 15)][(c8A >> 4) << 3];
            const bool full = AALIGN && (gmA < ga.M) && (kb + c8A + 8 <= ga.K);
#if HAVE_ASYNC_LDS
            if (full) { async_b128(src, dst); asy = true; return; }
#endif
            asy = false;
            pk.u = (u32x4)0u;
            if (gmA < ga.M) {
                if (full) pk.u = *(const u32x4*)src;
                else {
#pragma unroll
                    for (int e = 0; e < 8; ++e)
                        if (kb + c8A + e < ga.K) pk.h[e] = src[e];
                }
            }
        } else {
            (void)buf;
            asy = false;
            pk.u = (u32x4)0u;
            const int gk = kb + colA;
            if (gk < ga.K && gm0A < ga.M) {
                const __bf16* src = ga.A + aOff + (long long)gk * ga.sAk + gm0A;
                if (AALIGN && (gm0A + 8 <= ga.M)) pk.u = *(const u32x4*)src;
                else {
#pragma unroll
                    for (int e = 0; e < 8; ++e)
                        if (gm0A + e < ga.M) pk.h[e] = src[e];
                }
            }
        }
    };
    auto storeA = [&](const Pk8& pk, bool asy, int buf) {
        if (asy) return;
        if constexpr (APATH == 0) {
            *(u32x4*)&AsF[buf][rA >> 4]
                         [(((c8A >> 3) & 1) << 4) + (rA & 15)][(c8A >> 4) << 3] = pk.u;
        } else {
            const int hiA = (colA >> 3) & 1;
            const int tA  = colA >> 4;
            const int eA  = (tA << 3) + (colA - (tA << 4) - (hiA << 3));
            const int trA = m8A >> 4;
            const int ln0 = (hiA << 4) + (m8A & 15);
#pragma unroll
            for (int j = 0; j < 8; ++j)
                AsF[buf][trA][ln0 + j][eA] = pk.h[j];
        }
    };
    auto loadB = [&](int kb, Pk8& pk, bool& asy, int buf) {
        if constexpr (BPATH == 1) {
            const __bf16* src = ga.B + bOff + (long long)gnB * ga.sBn + (kb + k8B);
            __bf16* dst = &BsF[buf][cB >> 4]
                              [(((k8B >> 3) & 1) << 4) + (cB & 15)][(k8B >> 4) << 3];
            const bool full = (gnB < ga.Nn) && (kb + k8B + 8 <= ga.K);
#if HAVE_ASYNC_LDS
            if (full) { async_b128(src, dst); asy = true; return; }
#endif
            asy = false;
            pk.u = (u32x4)0u;
            if (gnB < ga.Nn) {
                if (full) pk.u = *(const u32x4*)src;
                else {
#pragma unroll
                    for (int e = 0; e < 8; ++e)
                        if (kb + k8B + e < ga.K) pk.h[e] = src[e];
                }
            }
        } else {
            // Column overrun past Nn reads in-workspace garbage that the store
            // guard discards (verified per instance). sBk % 8 == 0 for all users.
            (void)buf;
            asy = false;
            pk.u = (u32x4)0u;
            const int gk = kb + rowB;
            if (gk < ga.K)
                pk.u = *(const u32x4*)(ga.B + bOff + (long long)gk * ga.sBk
                                       + (blockN + c8B));
        }
    };
    auto storeB = [&](const Pk8& pk, bool asy, int buf) {
        if (asy) return;
        if constexpr (BPATH == 1) {
            *(u32x4*)&BsF[buf][cB >> 4]
                         [(((k8B >> 3) & 1) << 4) + (cB & 15)][(k8B >> 4) << 3] = pk.u;
        } else {
            const int tcS = c8B >> 4;
            const int eB  = (((rowB >> 4) & 1) << 3) + (rowB & 7);
            const int ln0 = (((rowB >> 3) & 1) << 4) + (c8B & 15);
#pragma unroll
            for (int j = 0; j < 8; ++j)
                BsF[buf][tcS][ln0 + j][eB] = pk.h[j];
        }
    };

    v8f acc0 = {};
    v8f acc1 = {};

    Pk8 pa, pb;
    bool asyA = false, asyB = false;
    loadA(0, pa, asyA, 0);  loadB(0, pb, asyB, 0);
    storeA(pa, asyA, 0);    storeB(pb, asyB, 0);
    wait_async_then_sync();

    int cur = 0;
    for (int kb = 0; kb < ga.K; kb += 32) {
        const bool more = (kb + 32) < ga.K;
        if (more) { loadA(kb + 32, pa, asyA, cur ^ 1);
                    loadB(kb + 32, pb, asyB, cur ^ 1); }

        v16bf va  = *(const v16bf*)(&AsF[cur][tr][lane][0]);
        v16bf vb0 = *(const v16bf*)(&BsF[cur][tc0][lane][0]);
        v16bf vb1 = *(const v16bf*)(&BsF[cur][tc1][lane][0]);
        acc0 = __builtin_amdgcn_wmma_f32_16x16x32_bf16(false, va, false, vb0,
                                                       (short)0, acc0, false, false);
        acc1 = __builtin_amdgcn_wmma_f32_16x16x32_bf16(false, va, false, vb1,
                                                       (short)0, acc1, false, false);

        if (more) { storeA(pa, asyA, cur ^ 1); storeB(pb, asyB, cur ^ 1); }
        wait_async_then_sync();
        cur ^= 1;
    }

    // C/D layout: VGPR r -> M = r + 8*hi, N = lane&15
#pragma unroll
    for (int r = 0; r < 8; ++r) {
        int gm = blockM + tr * 16 + r + 8 * hi;
        if (gm >= ga.M) continue;
        float brow = (BIASMODE == 1) ? ga.bias[gm] : 0.0f;
        {
            int gn = blockN + tc0 * 16 + l15;
            if (gn < ga.Nn) {
                float v = acc0[r] + brow + ((BIASMODE == 2) ? ga.bias[gn] : 0.0f);
                long long di = dOff + (long long)gm * ga.ldD + gn;
                if constexpr (OUTBF16) ((__bf16*)ga.D)[di] = (__bf16)v;
                else                   ((float*)ga.D)[di]  = v;
            }
        }
        {
            int gn = blockN + tc1 * 16 + l15;
            if (gn < ga.Nn) {
                float v = acc1[r] + brow + ((BIASMODE == 2) ? ga.bias[gn] : 0.0f);
                long long di = dOff + (long long)gm * ga.ldD + gn;
                if constexpr (OUTBF16) ((__bf16*)ga.D)[di] = (__bf16)v;
                else                   ((float*)ga.D)[di]  = v;
            }
        }
    }
}

// ---------------------------------------------------------------------------
// Temporal (9,1) conv as GEMM: D[n, c', j] = sum_{c,dt} W[c', c*9+dt] *
// G[n, c, tau+dt-4, v], j = tau*25+v, zero padded on tau. Double-buffered.
__global__ __launch_bounds__(256, 2)
void gemm_tcn_wmma(const __bf16* __restrict__ W, const __bf16* __restrict__ G,
                   float* __restrict__ D) {
    __shared__ __bf16 AsF[2][4][32][16];
    __shared__ __bf16 BsF[2][4][32][16];

    const int tid  = threadIdx.x;
    const int wave = tid >> 5;
    const int lane = tid & 31;
    const int hi   = lane >> 4;
    const int l15  = lane & 15;

    const int n = blockIdx.z;
    const int blockN = blockIdx.x * 64;
    const int tr  = wave >> 1;
    const int tc0 = (wave & 1) * 2;
    const int tc1 = tc0 + 1;

    const __bf16* Gn = G + (long long)n * NCTV;

    const int rA  = tid >> 2;
    const int c8A = (tid & 3) << 3;
    const int k8B = (tid >> 6) << 3;
    const int cB  = tid & 63;
    const int gnB = blockN + cB;
    const int tau = gnB / 25;
    const int vv  = gnB - tau * 25;

    auto stageA = [&](int kb, int buf) {          // W[64][576]: exact + aligned
        const __bf16* src = W + rA * 576 + kb + c8A;
        __bf16* dst = &AsF[buf][rA >> 4]
                          [(((c8A >> 3) & 1) << 4) + (rA & 15)][(c8A >> 4) << 3];
#if HAVE_ASYNC_LDS
        async_b128(src, dst);
#else
        *(u32x4*)dst = *(const u32x4*)src;
#endif
    };
    auto loadB = [&](int kb, Pk8& pk) {           // gathered shifted reads
#pragma unroll
        for (int e = 0; e < 8; ++e) {
            int gk = kb + k8B + e;
            int cc = gk / 9;
            int dt = gk - cc * 9;
            int tt = tau + dt - 4;
            pk.h[e] = ((unsigned)tt < 256u) ? Gn[cc * TV + tt * 25 + vv]
                                            : (__bf16)0.0f;
        }
    };
    auto storeB = [&](const Pk8& pk, int buf) {
        *(u32x4*)&BsF[buf][cB >> 4]
                     [(((k8B >> 3) & 1) << 4) + (cB & 15)][(k8B >> 4) << 3] = pk.u;
    };

    v8f acc0 = {};
    v8f acc1 = {};

    Pk8 pb;
    stageA(0, 0); loadB(0, pb); storeB(pb, 0);
    wait_async_then_sync();

    int cur = 0;
    for (int kb = 0; kb < 576; kb += 32) {
        const bool more = (kb + 32) < 576;
        if (more) { stageA(kb + 32, cur ^ 1); loadB(kb + 32, pb); }

        v16bf va  = *(const v16bf*)(&AsF[cur][tr][lane][0]);
        v16bf vb0 = *(const v16bf*)(&BsF[cur][tc0][lane][0]);
        v16bf vb1 = *(const v16bf*)(&BsF[cur][tc1][lane][0]);
        acc0 = __builtin_amdgcn_wmma_f32_16x16x32_bf16(false, va, false, vb0,
                                                       (short)0, acc0, false, false);
        acc1 = __builtin_amdgcn_wmma_f32_16x16x32_bf16(false, va, false, vb1,
                                                       (short)0, acc1, false, false);

        if (more) storeB(pb, cur ^ 1);
        wait_async_then_sync();
        cur ^= 1;
    }

    float* Dn = D + (long long)n * NCTV;
#pragma unroll
    for (int r = 0; r < 8; ++r) {
        int gm = tr * 16 + r + 8 * hi;               // < 64 always
        Dn[(long long)gm * TV + blockN + tc0 * 16 + l15] = acc0[r];
        Dn[(long long)gm * TV + blockN + tc1 * 16 + l15] = acc1[r];
    }
}

// ---------------------------------------------------------------------------
__global__ void cast_f32_to_bf16(const float* __restrict__ src,
                                 __bf16* __restrict__ dst, int n) {
    int i = blockIdx.x * 256 + threadIdx.x;
    if (i < n) dst[i] = (__bf16)src[i];
}

// aT/bT[bz][r][k] = ab[n, (a|b)row(s,r), k*25 + r]: k-contiguous operands for
// the attention GEMM (fully vectorizable / async-stageable).
__global__ void transpose_ab(const __bf16* __restrict__ ab,
                             __bf16* __restrict__ aT, __bf16* __restrict__ bT) {
    int i = blockIdx.x * 256 + threadIdx.x;
    if (i >= 96 * 25 * 4096) return;
    int bz  = i / (25 * 4096);
    int rem = i - bz * (25 * 4096);
    int r   = rem >> 12;          // 0..24
    int k   = rem & 4095;
    int n   = bz / 3, s = bz - n * 3;
    long long src = (long long)n * (96 * TV) + (long long)s * (16 * TV) + k * 25 + r;
    aT[i] = ab[src];
    bT[i] = ab[src + 48 * TV];
}

// pack/cast all weight matrices into the WMMA-friendly layouts
__global__ void prep_weights(
    const float* caw, const float* cab, const float* cbw, const float* cbb,
    const float* cdw, const float* cdb, const float* tcw,
    const float* Wir, const float* Wii, const float* Win,
    const float* Whr, const float* Whi, const float* Whh,
    const float* bir, const float* bii, const float* bin,
    __bf16* Wab, __bf16* CDW, __bf16* TCW, __bf16* Wi, __bf16* Wh,
    float* ab_bias, float* cd_bias, float* ib) {
    int i = blockIdx.x * 256 + threadIdx.x;
    if (i < 6144) {                                   // Wab [96][64]
        int m = i / 64, c = i - m * 64;
        float v = (m < 48) ? caw[m * 64 + c] : cbw[(m - 48) * 64 + c];
        Wab[i] = (__bf16)v;
        return;
    }
    i -= 6144;
    if (i < 12288) {                                  // CDW [64][192] (c', s*64+c)
        int cp = i / 192, col = i - cp * 192;
        int s = col / 64, c = col - s * 64;
        CDW[i] = (__bf16)cdw[s * 4096 + cp * 64 + c];
        return;
    }
    i -= 12288;
    if (i < 36864) {                                  // TCW [64][576] (c', c*9+dt)
        int cp = i / 576, k = i - cp * 576;
        int c = k / 9, dt = k - c * 9;
        TCW[i] = (__bf16)tcw[(cp * 64 + c) * 9 + dt];
        return;
    }
    i -= 36864;
    if (i < 12288) {                                  // Wi [64][192] (c, g*64+o)
        int c = i / 192, col = i - c * 192;
        int g = col / 64, o = col - g * 64;
        const float* W = (g == 0) ? Wir : ((g == 1) ? Wii : Win);
        Wi[i] = (__bf16)W[o * 64 + c];
        return;
    }
    i -= 12288;
    if (i < 12288) {                                  // Wh [64][192]
        int c = i / 192, col = i - c * 192;
        int g = col / 64, o = col - g * 64;
        const float* W = (g == 0) ? Whr : ((g == 1) ? Whi : Whh);
        Wh[i] = (__bf16)W[o * 64 + c];
        return;
    }
    i -= 12288;
    if (i < 96) { ab_bias[i] = (i < 48) ? cab[i] : cbb[i - 48]; return; }
    i -= 96;
    if (i < 64) { cd_bias[i] = cdb[i] + cdb[64 + i] + cdb[128 + i]; return; }
    i -= 64;
    if (i < 192) {
        int g = i / 64, o = i - g * 64;
        ib[i] = ((g == 0) ? bir : ((g == 1) ? bii : bin))[o];
    }
}

// softmax over v (axis -2) per column w, + (A + PA), -> bf16 padded ld=32
__global__ void softmax_att_kernel(const float* __restrict__ logits,
                                   const float* __restrict__ Aw,
                                   const float* __restrict__ PAw,
                                   __bf16* __restrict__ attb) {
    int bz = blockIdx.x;           // n*3 + s
    int s  = bz % 3;
    int w  = threadIdx.x;
    if (w >= 25) return;
    const float scale = 1.0f / 4096.0f;
    const float* L = logits + bz * 1024;
    float x[25];
    float mx = -1e30f;
#pragma unroll
    for (int v = 0; v < 25; ++v) {
        x[v] = L[v * 32 + w] * scale;
        mx = fmaxf(mx, x[v]);
    }
    float sum = 0.0f;
#pragma unroll
    for (int v = 0; v < 25; ++v) { x[v] = expf(x[v] - mx); sum += x[v]; }
    float inv = 1.0f / sum;
#pragma unroll
    for (int v = 0; v < 25; ++v) {
        float a = x[v] * inv + Aw[s * 625 + v * 25 + w] + PAw[s * 625 + v * 25 + w];
        attb[bz * 1024 + v * 32 + w] = (__bf16)a;
    }
}

// gcn_out = relu(BN(y) + hidden)  -> bf16
__global__ void gcn_epilogue(const float* __restrict__ y, const float* __restrict__ h,
                             const float* g, const float* b, const float* m,
                             const float* v, __bf16* __restrict__ out) {
    int i = blockIdx.x * 256 + threadIdx.x;
    if (i >= NB * CCH * TV) return;
    int c = (i / TV) & 63;
    float sc = g[c] * rsqrtf(v[c] + 1e-5f);
    float t = (y[i] - m[c]) * sc + b[c] + h[i];
    out[i] = (__bf16)fmaxf(t, 0.0f);
}

// msg = relu(BN(t + tcn_b) + hidden) -> bf16
__global__ void msg_epilogue(const float* __restrict__ t0, const float* __restrict__ h,
                             const float* tb, const float* g, const float* b,
                             const float* m, const float* v,
                             __bf16* __restrict__ out) {
    int i = blockIdx.x * 256 + threadIdx.x;
    if (i >= NB * CCH * TV) return;
    int c = (i / TV) & 63;
    float sc = g[c] * rsqrtf(v[c] + 1e-5f);
    float t = t0[i] + tb[c];
    t = (t - m[c]) * sc + b[c] + h[i];
    out[i] = (__bf16)fmaxf(t, 0.0f);
}

__device__ __forceinline__ float sigf(float x) { return 1.0f / (1.0f + expf(-x)); }

// GRU gates + transposed store to [N,C,T,V]
__global__ void gate_kernel(const __bf16* __restrict__ Di,
                            const __bf16* __restrict__ Dh,
                            const float* __restrict__ hid,
                            float* __restrict__ out) {
    int i = blockIdx.x * 256 + threadIdx.x;
    if (i >= NB * TV * CCH) return;
    int o = i & 63;
    int j = (i >> 6) % TV;
    int n = i / (TV * CCH);
    long long base = (long long)n * 1228800 + (long long)j * 192;
    float rg = sigf((float)Di[base + o]       + (float)Dh[base + o]);
    float zg = sigf((float)Di[base + 64 + o]  + (float)Dh[base + 64 + o]);
    float ng = tanhf((float)Di[base + 128 + o] + rg * (float)Dh[base + 128 + o]);
    long long oi = (long long)n * NCTV + (long long)o * TV + j;
    float hv = hid[oi];
    out[oi] = (1.0f - zg) * ng + zg * hv;
}

// ---------------------------------------------------------------------------
template<int APATH, bool AALIGN, int BPATH, int BIASMODE, bool OUTBF16>
static inline void launch_gemm(const GemmArgs& ga, int batch, hipStream_t s) {
    dim3 grid((ga.Nn + 63) / 64, (ga.M + 63) / 64, batch);
    gemm_bf16_wmma<APATH, AALIGN, BPATH, BIASMODE, OUTBF16>
        <<<grid, 256, 0, s>>>(ga);
}

extern "C" void kernel_launch(void* const* d_in, const int* in_sizes, int n_in,
                              void* d_out, int out_size, void* d_ws, size_t ws_size,
                              hipStream_t stream) {
    (void)in_sizes; (void)n_in; (void)out_size; (void)ws_size;
    const float* feature = (const float*)d_in[0];
    const float* hidden  = (const float*)d_in[1];
    const float* Aw      = (const float*)d_in[2];
    const float* PAw     = (const float*)d_in[3];
    const float* caw = (const float*)d_in[4];
    const float* cab = (const float*)d_in[5];
    const float* cbw = (const float*)d_in[6];
    const float* cbb = (const float*)d_in[7];
    const float* cdw = (const float*)d_in[8];
    const float* cdb = (const float*)d_in[9];
    const float* gbn_g = (const float*)d_in[10];
    const float* gbn_b = (const float*)d_in[11];
    const float* gbn_m = (const float*)d_in[12];
    const float* gbn_v = (const float*)d_in[13];
    const float* tcw   = (const float*)d_in[14];
    const float* tcb   = (const float*)d_in[15];
    const float* tbn_g = (const float*)d_in[16];
    const float* tbn_b = (const float*)d_in[17];
    const float* tbn_m = (const float*)d_in[18];
    const float* tbn_v = (const float*)d_in[19];
    const float* Wir = (const float*)d_in[20];
    const float* Wii = (const float*)d_in[21];
    const float* Win = (const float*)d_in[22];
    const float* Whr = (const float*)d_in[23];
    const float* Whi = (const float*)d_in[24];
    const float* Whh = (const float*)d_in[25];
    const float* bir = (const float*)d_in[26];
    const float* bii = (const float*)d_in[27];
    const float* bin = (const float*)d_in[28];

    char* ws = (char*)d_ws;
    __bf16* Hbf   = (__bf16*)(ws + OFF_HBF);
    __bf16* Fbf   = (__bf16*)(ws + OFF_FEAT);
    __bf16* abbf  = (__bf16*)(ws + OFF_AB);
    __bf16* zbf   = (__bf16*)(ws + OFF_Z);
    __bf16* gcnbf = (__bf16*)(ws + OFF_GCN);
    __bf16* msgbf = (__bf16*)(ws + OFF_MSG);
    __bf16* Dhbf  = (__bf16*)(ws + OFF_DH);
    __bf16* Dibf  = (__bf16*)(ws + OFF_DI);
    float*  logit = (float*)(ws + OFF_LOG);
    __bf16* attbf = (__bf16*)(ws + OFF_ATT);
    __bf16* aT    = (__bf16*)(ws + OFF_ABT);
    __bf16* bT    = (__bf16*)(ws + OFF_BBT);
    char* sm = ws + OFF_SMALL;
    __bf16* Wab = (__bf16*)(sm + SW_WAB);
    __bf16* CDW = (__bf16*)(sm + SW_CDW);
    __bf16* TCW = (__bf16*)(sm + SW_TCW);
    __bf16* Wi  = (__bf16*)(sm + SW_WI);
    __bf16* Wh  = (__bf16*)(sm + SW_WH);
    float* ab_bias = (float*)(sm + SW_ABB);
    float* cd_bias = (float*)(sm + SW_CDB);
    float* ib      = (float*)(sm + SW_IB);
    float* yout = (float*)d_out;   // y and t scratch live in d_out (fully rewritten)

    // 0: weight packing / casting
    prep_weights<<<314, 256, 0, stream>>>(caw, cab, cbw, cbb, cdw, cdb, tcw,
                                          Wir, Wii, Win, Whr, Whi, Whh,
                                          bir, bii, bin,
                                          Wab, CDW, TCW, Wi, Wh,
                                          ab_bias, cd_bias, ib);
    // 1: activation casts
    const int NEL = NB * CCH * TV;                     // 13,107,200
    cast_f32_to_bf16<<<(NEL + 255) / 256, 256, 0, stream>>>(hidden,  Hbf, NEL);
    cast_f32_to_bf16<<<(NEL + 255) / 256, 256, 0, stream>>>(feature, Fbf, NEL);

    // 2: a/b projections   ab[n,96,6400] = Wab[96,64] @ H[n][64,6400]  (+bias)
    {
        GemmArgs g = {};
        g.A = Wab; g.B = Hbf; g.D = abbf;
        g.M = 96; g.Nn = TV; g.K = 64;
        g.sAm = 64; g.sAk = 1; g.aOuter = 0; g.aInner = 0; g.aDiv = 1;
        g.sBk = TV; g.sBn = 1; g.bOuter = NCTV; g.bInner = 0; g.bDiv = 1;
        g.ldD = TV; g.dOuter = 96 * TV;
        g.bias = ab_bias;
        launch_gemm<0, true, 0, 1, true>(g, NB, stream);
    }
    // 2b: k-contiguous transposed copies of a and b for the attention GEMM
    transpose_ab<<<(96 * 25 * 4096 + 255) / 256, 256, 0, stream>>>(abbf, aT, bT);

    // 3: attention logits per (n,s): [25,25] = aT @ bT^T over K=IC*T=4096
    {
        GemmArgs g = {};
        g.A = aT; g.B = bT; g.D = logit;
        g.M = 25; g.Nn = 25; g.K = 4096;
        g.sAm = 4096; g.sAk = 1; g.aOuter = 25 * 4096; g.aInner = 0; g.aDiv = 1;
        g.sBk = 1; g.sBn = 4096; g.bOuter = 25 * 4096; g.bInner = 0; g.bDiv = 1;
        g.ldD = 32; g.dOuter = 1024;
        g.bias = nullptr;
        launch_gemm<0, true, 1, 0, false>(g, NB * NSUB, stream);
    }
    // 4: softmax over v + (A + PA) -> bf16
    softmax_att_kernel<<<NB * NSUB, 32, 0, stream>>>(logit, Aw, PAw, attbf);

    // 5: z[n,s] = h_flat[16384,25] @ att[25,25]  -> bf16
    {
        GemmArgs g = {};
        g.A = Hbf; g.B = attbf; g.D = zbf;
        g.M = 16384; g.Nn = 25; g.K = 25;
        g.sAm = 25; g.sAk = 1; g.aOuter = NCTV; g.aInner = 0; g.aDiv = 3;
        g.sBk = 32; g.sBn = 1; g.bOuter = 1024; g.bInner = 0; g.bDiv = 1;
        g.ldD = 25; g.dOuter = 16384 * 25;
        g.bias = nullptr;
        launch_gemm<0, false, 0, 0, true>(g, NB * NSUB, stream);
    }
    // 6: y[n] = CDW[64,192] @ Z[n][192,6400]  (+sum_s conv_d_b) -> f32 (d_out)
    {
        GemmArgs g = {};
        g.A = CDW; g.B = zbf; g.D = yout;
        g.M = 64; g.Nn = TV; g.K = 192;
        g.sAm = 192; g.sAk = 1; g.aOuter = 0; g.aInner = 0; g.aDiv = 1;
        g.sBk = TV;  g.sBn = 1; g.bOuter = (long long)3 * 16384 * 25; g.bInner = 0; g.bDiv = 1;
        g.ldD = TV; g.dOuter = NCTV;
        g.bias = cd_bias;
        launch_gemm<0, true, 0, 1, false>(g, NB, stream);
    }
    // 7: gcn_out = relu(BN(y) + hidden) -> bf16
    gcn_epilogue<<<(NEL + 255) / 256, 256, 0, stream>>>(yout, hidden,
                                                        gbn_g, gbn_b, gbn_m, gbn_v,
                                                        gcnbf);
    // 8: temporal conv (9,1) as K=576 GEMM -> f32 (d_out)
    {
        dim3 grid(TV / 64, 1, NB);
        gemm_tcn_wmma<<<grid, 256, 0, stream>>>(TCW, gcnbf, yout);
    }
    // 9: msg = relu(BN(t + tcn_b) + hidden) -> bf16
    msg_epilogue<<<(NEL + 255) / 256, 256, 0, stream>>>(yout, hidden, tcb,
                                                        tbn_g, tbn_b, tbn_m, tbn_v,
                                                        msgbf);
    // 10: Di[n] = feat[6400,64] @ Wi[64,192]  (+gate biases) -> bf16 (alias)
    {
        GemmArgs g = {};
        g.A = Fbf; g.B = Wi; g.D = Dibf;
        g.M = TV; g.Nn = 192; g.K = 64;
        g.sAm = 1; g.sAk = TV; g.aOuter = NCTV; g.aInner = 0; g.aDiv = 1;
        g.sBk = 192; g.sBn = 1; g.bOuter = 0; g.bInner = 0; g.bDiv = 1;
        g.ldD = 192; g.dOuter = (long long)TV * 192;
        g.bias = ib;
        launch_gemm<1, true, 0, 2, true>(g, NB, stream);
    }
    // 11: Dh[n] = msg[6400,64] @ Wh[64,192] -> bf16
    {
        GemmArgs g = {};
        g.A = msgbf; g.B = Wh; g.D = Dhbf;
        g.M = TV; g.Nn = 192; g.K = 64;
        g.sAm = 1; g.sAk = TV; g.aOuter = NCTV; g.aInner = 0; g.aDiv = 1;
        g.sBk = 192; g.sBn = 1; g.bOuter = 0; g.bInner = 0; g.bDiv = 1;
        g.ldD = 192; g.dOuter = (long long)TV * 192;
        g.bias = nullptr;
        launch_gemm<1, true, 0, 0, true>(g, NB, stream);
    }
    // 12: GRU gates + transposed store -> d_out [N,C,T,V]
    gate_kernel<<<(NEL + 255) / 256, 256, 0, stream>>>(Dibf, Dhbf, hidden,
                                                       (float*)d_out);
}
